// GNN_42253888258070
// MI455X (gfx1250) — matrix operand order repaired and checked by reading, not compiled, not run on hardware
//
#include <hip/hip_runtime.h>

#define N_NODES 50000
#define N_EDGES 600000
#define D_FEAT  128

typedef __attribute__((ext_vector_type(16))) __bf16 v16bf;
typedef __attribute__((ext_vector_type(8)))  __bf16 v8bf;
typedef __attribute__((ext_vector_type(8)))  float  v8f;

// ---------------------------------------------------------------------------
// WMMA helpers (wave32, 16x16x32 bf16 -> f32)
//
// A-frag (16x32, bf16): lane l holds row (l&15); elements 0..7 are K =
// kbase..kbase+7 and elements 8..15 are K = kbase+16..kbase+23 with
// kbase = (l<16 ? 0 : 8).  -> two 16B contiguous LDS loads per fragment.
//
// B-frag (32x16, bf16): lane l holds col (l&15); same K order as A-frag:
// element i is K = (l<16 ? 0 : 8) + (i&7) + (i<8 ? 0 : 16).  Weights are pre-packed so this is one
// contiguous 32B load per lane.
//
// C/D-frag (16x16, f32): VGPR r, lane l -> row r + (l<16?0:8), col (l&15).
// ---------------------------------------------------------------------------

__device__ __forceinline__ v16bf load_afrag(const __bf16* in, int din,
                                            int row, int ka, int kt) {
  v16bf a;
  const __bf16* ap = in + row * din + (kt << 5) + ka;
  ((v8bf*)&a)[0] = *(const v8bf*)(ap);
  ((v8bf*)&a)[1] = *(const v8bf*)(ap + 16);
  return a;
}

// Single 16-row tile (used by node-side kernels).
__device__ __forceinline__ v8f mlp_tile(const __bf16* in, int din,
                                        const __bf16* Wp, int nt, int NT,
                                        int lane) {
  v8f acc = {0.f, 0.f, 0.f, 0.f, 0.f, 0.f, 0.f, 0.f};
  const int row = lane & 15;
  const int ka  = (lane & 16) ? 8 : 0;
  #pragma unroll
  for (int kt = 0; kt < (din >> 5); ++kt) {
    v16bf a = load_afrag(in, din, row, ka, kt);
    v16bf b = *(const v16bf*)(Wp + (((size_t)(kt * NT + nt) << 5) + lane) * 16);
    acc = __builtin_amdgcn_wmma_f32_16x16x32_bf16(false, a, false, b,
                                                  (short)0, acc, false, false);
  }
  return acc;
}

// Two 16-row tiles sharing each B fragment (M=32 register blocking):
// one weight-fragment load now feeds two WMMAs.
__device__ __forceinline__ void mlp_tile2(const __bf16* in, int din,
                                          const __bf16* Wp, int nt, int NT,
                                          int lane, v8f& c0, v8f& c1) {
  v8f z = {0.f, 0.f, 0.f, 0.f, 0.f, 0.f, 0.f, 0.f};
  c0 = z; c1 = z;
  const int row = lane & 15;
  const int ka  = (lane & 16) ? 8 : 0;
  #pragma unroll
  for (int kt = 0; kt < (din >> 5); ++kt) {
    v16bf b = *(const v16bf*)(Wp + (((size_t)(kt * NT + nt) << 5) + lane) * 16);
    v16bf a0 = load_afrag(in, din, row,      ka, kt);
    v16bf a1 = load_afrag(in, din, row + 16, ka, kt);
    c0 = __builtin_amdgcn_wmma_f32_16x16x32_bf16(false, a0, false, b,
                                                 (short)0, c0, false, false);
    c1 = __builtin_amdgcn_wmma_f32_16x16x32_bf16(false, a1, false, b,
                                                 (short)0, c1, false, false);
  }
}

__device__ __forceinline__ void store_act(__bf16* out, int stride, int nt,
                                          v8f c, const float* bias, int lane) {
  const int col = (lane & 15) + (nt << 4);
  const int rb  = (lane & 16) ? 8 : 0;
  const float bv = bias[col];
  #pragma unroll
  for (int r = 0; r < 8; ++r) {
    float v = fmaxf(c[r] + bv, 0.0f);          // ReLU
    out[(rb + r) * stride + col] = (__bf16)v;
  }
}

__device__ __forceinline__ void store_act2(__bf16* out, int stride, int nt,
                                           v8f c0, v8f c1, const float* bias,
                                           int lane) {
  const int col = (lane & 15) + (nt << 4);
  const int rb  = (lane & 16) ? 8 : 0;
  const float bv = bias[col];
  #pragma unroll
  for (int r = 0; r < 8; ++r) {
    out[(rb + r) * stride + col]        = (__bf16)fmaxf(c0[r] + bv, 0.0f);
    out[(16 + rb + r) * stride + col]   = (__bf16)fmaxf(c1[r] + bv, 0.0f);
  }
}

// Layers 1 (din->256, ReLU) and 2 (256->128, ReLU), 16 rows.
template <int DIN>
__device__ __forceinline__ void mlp_front(const __bf16* in,
                                          const __bf16* pW1, const float* b1,
                                          const __bf16* pW2, const float* b2,
                                          __bf16* h1, __bf16* h2, int lane) {
  for (int nt = 0; nt < 16; ++nt) {
    v8f c = mlp_tile(in, DIN, pW1, nt, 16, lane);
    store_act(h1, 256, nt, c, b1, lane);
  }
  __syncthreads();
  for (int nt = 0; nt < 8; ++nt) {
    v8f c = mlp_tile(h1, 256, pW2, nt, 8, lane);
    store_act(h2, 128, nt, c, b2, lane);
  }
  __syncthreads();
}

// Layers 1 and 2, 32 rows (M=32 blocking).
template <int DIN>
__device__ __forceinline__ void mlp_front2(const __bf16* in,
                                           const __bf16* pW1, const float* b1,
                                           const __bf16* pW2, const float* b2,
                                           __bf16* h1, __bf16* h2, int lane) {
  for (int nt = 0; nt < 16; ++nt) {
    v8f c0, c1;
    mlp_tile2(in, DIN, pW1, nt, 16, lane, c0, c1);
    store_act2(h1, 256, nt, c0, c1, b1, lane);
  }
  __syncthreads();
  for (int nt = 0; nt < 8; ++nt) {
    v8f c0, c1;
    mlp_tile2(h1, 256, pW2, nt, 8, lane, c0, c1);
    store_act2(h2, 128, nt, c0, c1, b2, lane);
  }
  __syncthreads();
}

// ---------------------------------------------------------------------------
// Weight packing: fp32 row-major [K,N] -> bf16 B-fragment layout
// ---------------------------------------------------------------------------
__global__ void pack_w_kernel(const float* __restrict__ W,
                              __bf16* __restrict__ out, int K, int N) {
  int t = blockIdx.x * blockDim.x + threadIdx.x;
  if (t >= K * N) return;
  int i    = t & 15;
  int lane = (t >> 4) & 31;
  int tile = t >> 9;
  int NT = N >> 4;
  int nt = tile % NT;
  int kt = tile / NT;
  int k = (kt << 5) + ((lane & 16) ? 8 : 0) + (i & 7) + ((i & 8) ? 16 : 0);
  int n = (nt << 4) + (lane & 15);
  out[t] = (__bf16)W[(size_t)k * N + n];
}

__global__ void fill0_kernel(float* __restrict__ p, int n) {
  for (int i = blockIdx.x * blockDim.x + threadIdx.x; i < n;
       i += gridDim.x * blockDim.x)
    p[i] = 0.0f;
}

__global__ void deg_kernel(const int* __restrict__ dst,
                           float* __restrict__ deg, int E) {
  int t = blockIdx.x * blockDim.x + threadIdx.x;
  if (t < E) atomicAdd(&deg[dst[t]], 1.0f);
}

// ---------------------------------------------------------------------------
// h_v = MLP_v(x_v): 128 -> 256 -> 128 -> 128. One wave per 16 nodes.
// ---------------------------------------------------------------------------
__global__ void __launch_bounds__(32)
node_v_kernel(const float* __restrict__ x,
              const __bf16* pW1, const __bf16* pW2, const __bf16* pW3,
              const float* b1, const float* b2, const float* b3,
              float* __restrict__ hvf, __bf16* __restrict__ hvb) {
  __shared__ __align__(16) __bf16 s_in[16 * 128];
  __shared__ __align__(16) __bf16 s_h1[16 * 256];
  __shared__ __align__(16) __bf16 s_h2[16 * 128];
  const int lane = threadIdx.x;
  const int row0 = blockIdx.x * 16;
  // stage 16x128 f32 -> bf16
  for (int q = lane; q < 16 * 32; q += 32) {
    int r = q >> 5, c4 = q & 31;
    float4 v = ((const float4*)(x + (size_t)(row0 + r) * 128))[c4];
    __bf16* d = s_in + r * 128 + c4 * 4;
    d[0] = (__bf16)v.x; d[1] = (__bf16)v.y;
    d[2] = (__bf16)v.z; d[3] = (__bf16)v.w;
  }
  __syncthreads();
  mlp_front<128>(s_in, pW1, b1, pW2, b2, s_h1, s_h2, lane);
  for (int nt = 0; nt < 8; ++nt) {
    v8f c = mlp_tile(s_h2, 128, pW3, nt, 8, lane);
    const int col = (lane & 15) + (nt << 4);
    const int rb  = (lane & 16) ? 8 : 0;
    const float bv = b3[col];
    #pragma unroll
    for (int r = 0; r < 8; ++r) {
      float v = c[r] + bv;
      size_t off = (size_t)(row0 + rb + r) * 128 + col;
      hvf[off] = v;
      hvb[off] = (__bf16)v;
    }
  }
}

// ---------------------------------------------------------------------------
// h_e = MLP_e(l_e): 1 -> 256 (rank-1, VALU) -> 128 -> 128 (WMMA, M=32).
// ---------------------------------------------------------------------------
__global__ void __launch_bounds__(32)
edge_e_kernel(const float* __restrict__ le,
              const float* __restrict__ W1, const float* __restrict__ b1,
              const __bf16* pW2, const __bf16* pW3,
              const float* b2, const float* b3,
              __bf16* __restrict__ he) {
  __shared__ __align__(16) __bf16 s_h1[32 * 256];
  __shared__ __align__(16) __bf16 s_h2[32 * 128];
  __shared__ float s_le[32];
  const int lane = threadIdx.x;
  const int e0 = blockIdx.x * 32;
  s_le[lane] = le[e0 + lane];
  __syncthreads();
  for (int q = lane; q < 32 * 256; q += 32) {
    int r = q >> 8, c = q & 255;
    s_h1[q] = (__bf16)fmaxf(s_le[r] * W1[c] + b1[c], 0.0f);
  }
  __syncthreads();
  for (int nt = 0; nt < 8; ++nt) {
    v8f c0, c1;
    mlp_tile2(s_h1, 256, pW2, nt, 8, lane, c0, c1);
    store_act2(s_h2, 128, nt, c0, c1, b2, lane);
  }
  __syncthreads();
  for (int nt = 0; nt < 8; ++nt) {
    v8f c0, c1;
    mlp_tile2(s_h2, 128, pW3, nt, 8, lane, c0, c1);
    const int col = (lane & 15) + (nt << 4);
    const int rb  = (lane & 16) ? 8 : 0;
    const float bv = b3[col];
    #pragma unroll
    for (int r = 0; r < 8; ++r) {
      he[(size_t)(e0 + rb + r) * 128 + col]      = (__bf16)(c0[r] + bv);
      he[(size_t)(e0 + 16 + rb + r) * 128 + col] = (__bf16)(c1[r] + bv);
    }
  }
}

// ---------------------------------------------------------------------------
// Fused edge message MLPs (384->256->128->128 twice), M=32 per wave,
// atomic segment-sum scatter into sum1/sum0[dst].
// ---------------------------------------------------------------------------
__global__ void __launch_bounds__(32)
edge_msg_kernel(const int* __restrict__ src, const int* __restrict__ dst,
                const __bf16* __restrict__ hvb, const __bf16* __restrict__ he,
                const __bf16* p1W1, const __bf16* p1W2, const __bf16* p1W3,
                const float* b11, const float* b12, const float* b13,
                const __bf16* p0W1, const __bf16* p0W2, const __bf16* p0W3,
                const float* b01, const float* b02, const float* b03,
                float* __restrict__ sum1, float* __restrict__ sum0) {
  __shared__ __align__(16) __bf16 s_in[32 * 384];
  __shared__ __align__(16) __bf16 s_h1[32 * 256];
  __shared__ __align__(16) __bf16 s_h2[32 * 128];
  __shared__ int s_src[32], s_dst[32];
  const int lane = threadIdx.x;
  const int e0 = blockIdx.x * 32;
  s_src[lane] = src[e0 + lane];
  s_dst[lane] = dst[e0 + lane];
  __syncthreads();
  // gather edge_in = [h_v[src] | h_v[dst] | h_e] : 32 rows x 48 x uint4
  for (int q = lane; q < 32 * 48; q += 32) {
    int r = q / 48, c = q % 48;
    const uint4* sp;
    if (c < 16)       sp = (const uint4*)(hvb + (size_t)s_src[r] * 128) + c;
    else if (c < 32)  sp = (const uint4*)(hvb + (size_t)s_dst[r] * 128) + (c - 16);
    else              sp = (const uint4*)(he  + (size_t)(e0 + r) * 128) + (c - 32);
    ((uint4*)s_in)[r * 48 + c] = *sp;
  }
  __syncthreads();
  // ---- m1 = MLP_edge1(edge_in), scattered into sum1[dst] ----
  mlp_front2<384>(s_in, p1W1, b11, p1W2, b12, s_h1, s_h2, lane);
  for (int nt = 0; nt < 8; ++nt) {
    v8f c0, c1;
    mlp_tile2(s_h2, 128, p1W3, nt, 8, lane, c0, c1);
    const int col = (lane & 15) + (nt << 4);
    const int rb  = (lane & 16) ? 8 : 0;
    const float bv = b13[col];
    #pragma unroll
    for (int r = 0; r < 8; ++r) {
      atomicAdd(&sum1[(size_t)s_dst[rb + r] * 128 + col],      c0[r] + bv);
      atomicAdd(&sum1[(size_t)s_dst[16 + rb + r] * 128 + col], c1[r] + bv);
    }
  }
  __syncthreads();
  // ---- m0 = MLP_edge0(edge_in), scattered into sum0[dst] ----
  mlp_front2<384>(s_in, p0W1, b01, p0W2, b02, s_h1, s_h2, lane);
  for (int nt = 0; nt < 8; ++nt) {
    v8f c0, c1;
    mlp_tile2(s_h2, 128, p0W3, nt, 8, lane, c0, c1);
    const int col = (lane & 15) + (nt << 4);
    const int rb  = (lane & 16) ? 8 : 0;
    const float bv = b03[col];
    #pragma unroll
    for (int r = 0; r < 8; ++r) {
      atomicAdd(&sum0[(size_t)s_dst[rb + r] * 128 + col],      c0[r] + bv);
      atomicAdd(&sum0[(size_t)s_dst[16 + rb + r] * 128 + col], c1[r] + bv);
    }
  }
}

// ---------------------------------------------------------------------------
// Aggregation: upd_in=[h_v | mean m1 | mean m0], MLP_aggr, +h_v, where(deg>0)
// ---------------------------------------------------------------------------
__global__ void __launch_bounds__(32)
aggr_kernel(const __bf16* __restrict__ hvb, const float* __restrict__ hvf,
            const float* __restrict__ sum1, const float* __restrict__ sum0,
            const float* __restrict__ deg,
            const __bf16* pW1, const __bf16* pW2, const __bf16* pW3,
            const float* b1, const float* b2, const float* b3,
            float* __restrict__ out) {
  __shared__ __align__(16) __bf16 s_in[16 * 384];
  __shared__ __align__(16) __bf16 s_h1[16 * 256];
  __shared__ __align__(16) __bf16 s_h2[16 * 128];
  __shared__ float s_deg[16], s_rdeg[16];
  const int lane = threadIdx.x;
  const int n0 = blockIdx.x * 16;
  if (lane < 16) {
    float d = deg[n0 + lane];
    s_deg[lane]  = d;
    s_rdeg[lane] = 1.0f / fmaxf(d, 1.0f);
  }
  __syncthreads();
  for (int q = lane; q < 16 * 384; q += 32) {
    int r = q / 384, c = q % 384;
    __bf16 v;
    if (c < 128)      v = hvb[(size_t)(n0 + r) * 128 + c];
    else if (c < 256) v = (__bf16)(sum1[(size_t)(n0 + r) * 128 + (c - 128)] * s_rdeg[r]);
    else              v = (__bf16)(sum0[(size_t)(n0 + r) * 128 + (c - 256)] * s_rdeg[r]);
    s_in[q] = v;
  }
  __syncthreads();
  mlp_front<384>(s_in, pW1, b1, pW2, b2, s_h1, s_h2, lane);
  for (int nt = 0; nt < 8; ++nt) {
    v8f c = mlp_tile(s_h2, 128, pW3, nt, 8, lane);
    const int col = (lane & 15) + (nt << 4);
    const int rb  = (lane & 16) ? 8 : 0;
    const float bv = b3[col];
    #pragma unroll
    for (int r = 0; r < 8; ++r) {
      int row = rb + r;
      size_t off = (size_t)(n0 + row) * 128 + col;
      float hv = hvf[off];
      float vn = c[r] + bv + hv;                 // residual
      out[off] = (s_deg[row] > 0.0f) ? vn : hv;  // DGL recv semantics
    }
  }
}

// ---------------------------------------------------------------------------
extern "C" void kernel_launch(void* const* d_in, const int* in_sizes, int n_in,
                              void* d_out, int out_size, void* d_ws,
                              size_t ws_size, hipStream_t stream) {
  (void)in_sizes; (void)n_in; (void)out_size; (void)ws_size;
  const float* x_v = (const float*)d_in[0];
  const float* l_e = (const float*)d_in[1];
  const int*   src = (const int*)d_in[2];
  const int*   dst = (const int*)d_in[3];
  // params flattened: dicts sorted by key -> aggr, e, edge0, edge1, v;
  // within each: W1, W2, W3, b1, b2, b3.
  const float* aW1 = (const float*)d_in[4];
  const float* aW2 = (const float*)d_in[5];
  const float* aW3 = (const float*)d_in[6];
  const float* ab1 = (const float*)d_in[7];
  const float* ab2 = (const float*)d_in[8];
  const float* ab3 = (const float*)d_in[9];
  const float* eW1 = (const float*)d_in[10];
  const float* eW2 = (const float*)d_in[11];
  const float* eW3 = (const float*)d_in[12];
  const float* eb1 = (const float*)d_in[13];
  const float* eb2 = (const float*)d_in[14];
  const float* eb3 = (const float*)d_in[15];
  const float* g0W1 = (const float*)d_in[16];
  const float* g0W2 = (const float*)d_in[17];
  const float* g0W3 = (const float*)d_in[18];
  const float* g0b1 = (const float*)d_in[19];
  const float* g0b2 = (const float*)d_in[20];
  const float* g0b3 = (const float*)d_in[21];
  const float* g1W1 = (const float*)d_in[22];
  const float* g1W2 = (const float*)d_in[23];
  const float* g1W3 = (const float*)d_in[24];
  const float* g1b1 = (const float*)d_in[25];
  const float* g1b2 = (const float*)d_in[26];
  const float* g1b3 = (const float*)d_in[27];
  const float* vW1 = (const float*)d_in[28];
  const float* vW2 = (const float*)d_in[29];
  const float* vW3 = (const float*)d_in[30];
  const float* vb1 = (const float*)d_in[31];
  const float* vb2 = (const float*)d_in[32];
  const float* vb3 = (const float*)d_in[33];

  char* ws = (char*)d_ws;
  float*  hvf  = (float*)(ws + 0);            // 50000*128 f32  (25.6 MB)
  float*  sum1 = (float*)(ws + 25600000);     // 50000*128 f32
  float*  sum0 = (float*)(ws + 51200000);     // 50000*128 f32
  float*  degp = (float*)(ws + 76800000);     // 50000 f32
  __bf16* hvb  = (__bf16*)(ws + 77000000);    // 50000*128 bf16
  __bf16* he   = (__bf16*)(ws + 89800000);    // 600000*128 bf16 (153.6 MB)
  __bf16* pk   = (__bf16*)(ws + 243400000);   // packed weights (~1.1 MB)
  __bf16 *pV1 = pk,          *pV2 = pk + 32768,  *pV3 = pk + 65536;
  __bf16 *pE2 = pk + 81920,  *pE3 = pk + 114688;
  __bf16 *p1W1 = pk + 131072, *p1W2 = pk + 229376, *p1W3 = pk + 262144;
  __bf16 *p0W1 = pk + 278528, *p0W2 = pk + 376832, *p0W3 = pk + 409600;
  __bf16 *pA1 = pk + 425984,  *pA2 = pk + 524288,  *pA3 = pk + 557056;

  auto packw = [&](const float* W, __bf16* o, int K, int N) {
    int tot = K * N;
    pack_w_kernel<<<(tot + 255) / 256, 256, 0, stream>>>(W, o, K, N);
  };
  packw(vW1,  pV1,  128, 256); packw(vW2,  pV2,  256, 128); packw(vW3,  pV3,  128, 128);
  packw(eW2,  pE2,  256, 128); packw(eW3,  pE3,  128, 128);
  packw(g1W1, p1W1, 384, 256); packw(g1W2, p1W2, 256, 128); packw(g1W3, p1W3, 128, 128);
  packw(g0W1, p0W1, 384, 256); packw(g0W2, p0W2, 256, 128); packw(g0W3, p0W3, 128, 128);
  packw(aW1,  pA1,  384, 256); packw(aW2,  pA2,  256, 128); packw(aW3,  pA3,  128, 128);

  // zero sum1|sum0|deg (contiguous region)
  fill0_kernel<<<4096, 256, 0, stream>>>(sum1, (2 * N_NODES * D_FEAT) + N_NODES);

  node_v_kernel<<<N_NODES / 16, 32, 0, stream>>>(x_v, pV1, pV2, pV3,
                                                 vb1, vb2, vb3, hvf, hvb);
  edge_e_kernel<<<N_EDGES / 32, 32, 0, stream>>>(l_e, eW1, eb1, pE2, pE3,
                                                 eb2, eb3, he);
  deg_kernel<<<(N_EDGES + 255) / 256, 256, 0, stream>>>(dst, degp, N_EDGES);
  edge_msg_kernel<<<N_EDGES / 32, 32, 0, stream>>>(
      src, dst, hvb, he,
      p1W1, p1W2, p1W3, g1b1, g1b2, g1b3,
      p0W1, p0W2, p0W3, g0b1, g0b2, g0b3,
      sum1, sum0);
  aggr_kernel<<<N_NODES / 16, 32, 0, stream>>>(hvb, hvf, sum1, sum0, degp,
                                               pA1, pA2, pA3, ab1, ab2, ab3,
                                               (float*)d_out);
}
